// SpatialTransformer_27376121545457
// MI455X (gfx1250) — matrix operand order repaired and checked by reading, not compiled, not run on hardware
//
#include <hip/hip_runtime.h>
#include <hip/hip_bf16.h>

typedef __attribute__((ext_vector_type(16))) __bf16        v16bf;
typedef __attribute__((ext_vector_type(8)))  float         v8f;
typedef __attribute__((ext_vector_type(8)))  unsigned int  v8u;
typedef int v4i __attribute__((vector_size(4 * sizeof(int))));

union Frag { v8u u; v16bf h; };

// load 8 dwords (32B) from two contiguous 16B spans
#define LOAD_PAIR(frag, p0, p1)                                   \
    do {                                                          \
        ((uint4*)&(frag).u)[0] = *(const uint4*)(p0);             \
        ((uint4*)&(frag).u)[1] = *(const uint4*)(p1);             \
    } while (0)

// ---- optional CDNA5 async global->LDS path (guarded; fallback = plain copy)
#if defined(__has_builtin)
#if __has_builtin(__builtin_amdgcn_global_load_async_to_lds_b128)
#define HAVE_ASYNC_LDS 1
#endif
#endif
#ifndef HAVE_ASYNC_LDS
#define HAVE_ASYNC_LDS 0
#endif
#define GPTR(p) ((__attribute__((address_space(1))) v4i*)(p))
#define LPTR(p) ((__attribute__((address_space(3))) v4i*)(p))

__device__ __forceinline__ void wait_async_all() {
#if HAVE_ASYNC_LDS
#if __has_builtin(__builtin_amdgcn_s_wait_asynccnt)
    __builtin_amdgcn_s_wait_asynccnt(0);
#else
    asm volatile("s_wait_asynccnt 0x0" ::: "memory");
#endif
#endif
}

// ---------------------------------------------------------------------------
// f32 -> bf16 conversion (grid-stride), plain row-major
// ---------------------------------------------------------------------------
__global__ void k_conv(const float* __restrict__ in, __hip_bfloat16* __restrict__ out, long n) {
    long i = (long)blockIdx.x * blockDim.x + threadIdx.x;
    long stride = (long)gridDim.x * blockDim.x;
    for (; i < n; i += stride) out[i] = __float2bfloat16(in[i]);
}

// f32 (K,N) row-major -> bf16 in WMMA B-fragment block order.
__global__ void k_pack(const float* __restrict__ in, __hip_bfloat16* __restrict__ out,
                       int K, int N) {
    long total = (long)K * N;
    long i = (long)blockIdx.x * blockDim.x + threadIdx.x;
    long stride = (long)gridDim.x * blockDim.x;
    for (; i < total; i += stride) {
        int k = (int)(i / N), n = (int)(i % N);
        long blk = (long)(k >> 5) * (N >> 4) + (n >> 4);
        long idx = (blk << 9) + (((k & 16) + (n & 15)) << 4) + (k & 15);
        out[idx] = __float2bfloat16(in[i]);
    }
}

__global__ void k_zero_bf16(__hip_bfloat16* __restrict__ p, long n) {
    long i = (long)blockIdx.x * blockDim.x + threadIdx.x;
    long stride = (long)gridDim.x * blockDim.x;
    for (; i < n; i += stride) ((unsigned short*)p)[i] = 0;
}

// ---------------------------------------------------------------------------
// Coalesced LDS-tiled transpose: in (R x C) -> out (C x R), batched over z.
// Block 256 thr handles a 32x32 tile.
// ---------------------------------------------------------------------------
__global__ __launch_bounds__(256) void k_transpose(const float* __restrict__ in,
                                                   float* __restrict__ out,
                                                   int R, int C) {
    __shared__ float tile[32][33];
    long ib = (long)blockIdx.z * R * C;
    int r0 = blockIdx.y * 32, c0 = blockIdx.x * 32;
    int tx = threadIdx.x & 31, ty = threadIdx.x >> 5;   // 32 x 8
#pragma unroll
    for (int i = 0; i < 32; i += 8)
        tile[ty + i][tx] = in[ib + (long)(r0 + ty + i) * C + c0 + tx];
    __syncthreads();
#pragma unroll
    for (int i = 0; i < 32; i += 8)
        out[ib + (long)(c0 + ty + i) * R + r0 + tx] = tile[tx][ty + i];
}

// ---------------------------------------------------------------------------
// t = x + LN(x), rows of 512, f32 in -> f32 out.  One wave per row.
// ---------------------------------------------------------------------------
__global__ __launch_bounds__(256) void k_ln1f(const float* __restrict__ X,
                                              const float* __restrict__ g,
                                              const float* __restrict__ bb,
                                              float* __restrict__ T) {
    int lane = threadIdx.x & 31, wid = threadIdx.x >> 5;
    long row = (long)blockIdx.x * 8 + wid;
    const float* rp = X + row * 512;
    float vals[16];
    float s = 0.f;
#pragma unroll
    for (int i = 0; i < 16; i++) { vals[i] = rp[lane + i * 32]; s += vals[i]; }
#pragma unroll
    for (int m = 1; m < 32; m <<= 1) s += __shfl_xor(s, m, 32);
    float mu = s * (1.0f / 512.0f);
    float vs = 0.f;
#pragma unroll
    for (int i = 0; i < 16; i++) { float d = vals[i] - mu; vs += d * d; }
#pragma unroll
    for (int m = 1; m < 32; m <<= 1) vs += __shfl_xor(vs, m, 32);
    float inv = rsqrtf(vs * (1.0f / 512.0f) + 1e-5f);
    float* tp = T + row * 512;
#pragma unroll
    for (int i = 0; i < 16; i++) {
        int c = lane + i * 32;
        tp[c] = vals[i] + (vals[i] - mu) * inv * g[c] + bb[c];
    }
}

// LN: t (rows, 512) f32 -> bf16.  One wave per row.
__global__ __launch_bounds__(256) void k_ln(const float* __restrict__ T,
                                            const float* __restrict__ g,
                                            const float* __restrict__ bb,
                                            __hip_bfloat16* __restrict__ out) {
    int lane = threadIdx.x & 31, wid = threadIdx.x >> 5;
    long row = (long)blockIdx.x * 8 + wid;
    const float* rp = T + row * 512;
    float vals[16];
    float s = 0.f;
#pragma unroll
    for (int i = 0; i < 16; i++) { vals[i] = rp[lane + i * 32]; s += vals[i]; }
#pragma unroll
    for (int m = 1; m < 32; m <<= 1) s += __shfl_xor(s, m, 32);
    float mu = s * (1.0f / 512.0f);
    float vs = 0.f;
#pragma unroll
    for (int i = 0; i < 16; i++) { float d = vals[i] - mu; vs += d * d; }
#pragma unroll
    for (int m = 1; m < 32; m <<= 1) vs += __shfl_xor(vs, m, 32);
    float inv = rsqrtf(vs * (1.0f / 512.0f) + 1e-5f);
    __hip_bfloat16* op = out + row * 512;
#pragma unroll
    for (int i = 0; i < 16; i++) {
        int c = lane + i * 32;
        op[c] = __float2bfloat16((vals[i] - mu) * inv * g[c] + bb[c]);
    }
}

// ---------------------------------------------------------------------------
// Batched bf16 WMMA GEMM: C = act(A @ W + bias)   [optionally += resid]
//   A: (M,K) bf16 row-major;  W: fragment-packed (see k_pack); K % 32 == 0
//   mode 1: bf16 store; 2: exact GELU->bf16; 3: resid f32 +=; 4: store packed
// Block = 256 thr (8 waves); wave computes 32 rows x 64 cols; blk = 256x64.
// Row guards replaced by clamped loads (store side still guarded).
// ---------------------------------------------------------------------------
__global__ __launch_bounds__(256)
void k_gemm(const __hip_bfloat16* __restrict__ A, int lda, long strideA,
            const __hip_bfloat16* __restrict__ W,
            const float* __restrict__ bias,
            int M, int N, int K,
            __hip_bfloat16* __restrict__ outb, float* __restrict__ resid,
            int ldc, long strideC, int mode) {
    int lane = threadIdx.x & 31, wid = threadIdx.x >> 5;
    A += (long)blockIdx.z * strideA;
    long coff = (long)blockIdx.z * strideC;

    int rowBase = blockIdx.x * 256 + wid * 32;
    int colBase = blockIdx.y * 64;
    int m    = lane & 15;
    int sel8 = (lane >> 4) << 3;

    int r0 = min(rowBase + m,      M - 1);
    int r1 = min(rowBase + 16 + m, M - 1);

    const __hip_bfloat16* Ap0 = A + (long)r0 * lda + (sel8 << 1);
    const __hip_bfloat16* Ap1 = A + (long)r1 * lda + (sel8 << 1);
    const __hip_bfloat16* Wp  = W + ((long)(colBase >> 4) << 9) + (lane << 4);
    long wstep = (long)(N >> 4) << 9;

    v8f acc[2][4];
#pragma unroll
    for (int g = 0; g < 2; g++)
#pragma unroll
        for (int j = 0; j < 4; j++)
#pragma unroll
            for (int e = 0; e < 8; e++) acc[g][j][e] = 0.f;

    for (int kk = 0; kk < K; kk += 32) {
        Frag a0, a1, bfr[4];
        LOAD_PAIR(a0, Ap0, Ap0 + 8);
        LOAD_PAIR(a1, Ap1, Ap1 + 8);
#pragma unroll
        for (int j = 0; j < 4; j++)
            LOAD_PAIR(bfr[j], Wp + j * 512, Wp + j * 512 + 8);
        __builtin_prefetch(Ap0 + 64, 0, 1);
        __builtin_prefetch(Ap1 + 64, 0, 1);
#pragma unroll
        for (int j = 0; j < 4; j++) {
            acc[0][j] = __builtin_amdgcn_wmma_f32_16x16x32_bf16(
                false, a0.h, false, bfr[j].h, (short)0, acc[0][j], false, false);
            acc[1][j] = __builtin_amdgcn_wmma_f32_16x16x32_bf16(
                false, a1.h, false, bfr[j].h, (short)0, acc[1][j], false, false);
        }
        Ap0 += 32; Ap1 += 32; Wp += wstep;
    }

    int n = lane & 15;
#pragma unroll
    for (int g = 0; g < 2; g++) {
        int rowOut = rowBase + g * 16 + sel8;
#pragma unroll
        for (int j = 0; j < 4; j++) {
            int col = colBase + j * 16 + n;
            float bv = bias ? bias[col] : 0.f;
#pragma unroll
            for (int r = 0; r < 8; r++) {
                int rr = rowOut + r;
                if (rr < M) {
                    float val = acc[g][j][r] + bv;
                    if (mode == 2) val = 0.5f * val * (1.0f + erff(val * 0.70710678f));
                    if (mode == 3) {
                        resid[coff + (long)rr * ldc + col] += val;
                    } else if (mode == 4) {
                        long blk = (long)(rr >> 5) * (N >> 4) + (col >> 4);
                        long idx = coff + (blk << 9) + (((rr & 16) + (col & 15)) << 4) + (rr & 15);
                        outb[idx] = __float2bfloat16(val);
                    } else {
                        outb[coff + (long)rr * ldc + col] = __float2bfloat16(val);
                    }
                }
            }
        }
    }
}

// ---------------------------------------------------------------------------
// Multi-head cross attention, WMMA everywhere.
//   Q: (B*4096, 512) bf16 row-major; Kp: (B, Spad, 512) row-major zero-padded;
//   Vp: (B, Spad, 512) fragment-packed (k_gemm mode 4); O: (B*4096,512) bf16.
// Whole block shares (b,h): the K tile (Spad x 64) is staged into LDS once per
// block via async global->LDS (when available), then fragments are ds_load'ed.
// ---------------------------------------------------------------------------
template <int ST>
__global__ __launch_bounds__(256)
void k_attn(const __hip_bfloat16* __restrict__ Q,
            const __hip_bfloat16* __restrict__ Kp,
            const __hip_bfloat16* __restrict__ Vp,
            __hip_bfloat16* __restrict__ O,
            int Sact, float scale) {
    constexpr int Spad = ST * 16;
    __shared__ __align__(16) __hip_bfloat16 ksh[Spad][72];   // padded rows (144B)
    __shared__ __align__(16) __hip_bfloat16 pst[8][16][40];  // staged P chunk (16x32)

    int lane = threadIdx.x & 31, wid = threadIdx.x >> 5;
    int h = blockIdx.y, b = blockIdx.z;
    int tile = blockIdx.x * 8 + wid;            // 0..255
    int n0 = tile * 16;
    int m    = lane & 15;
    int sel8 = (lane >> 4) << 3;
    int kb   = (lane >> 4) << 4;

    // ---- cooperative stage of K (Spad x 64) for this (b,h) into LDS ----
    const __hip_bfloat16* kbase = Kp + (long)b * Spad * 512 + h * 64;
    for (int c = threadIdx.x; c < Spad * 8; c += 256) {
        int row = c >> 3, cc = (c & 7) << 3;    // 8 bf16 per 16B chunk
        const __hip_bfloat16* gp = kbase + (long)row * 512 + cc;
#if HAVE_ASYNC_LDS
        __builtin_amdgcn_global_load_async_to_lds_b128(GPTR(gp), LPTR(&ksh[row][cc]), 0, 0);
#else
        *(uint4*)&ksh[row][cc] = *(const uint4*)gp;
#endif
    }
    wait_async_all();
    __syncthreads();

    // Q fragments for d=64 (two K-steps of 32)
    Frag qf[2];
    const __hip_bfloat16* qp = Q + ((long)b * 4096 + n0 + m) * 512 + h * 64;
#pragma unroll
    for (int t = 0; t < 2; t++)
        LOAD_PAIR(qf[t], qp + t * 32 + kb, qp + t * 32 + kb + 8);

    // sim = Q K^T  (K^T fragments straight out of LDS)
    v8f sim[ST];
#pragma unroll
    for (int st = 0; st < ST; st++)
#pragma unroll
        for (int e = 0; e < 8; e++) sim[st][e] = 0.f;

#pragma unroll
    for (int st = 0; st < ST; st++) {
#pragma unroll
        for (int t = 0; t < 2; t++) {
            Frag bf;
            const __hip_bfloat16* kp = &ksh[st * 16 + m][t * 32 + kb];
            LOAD_PAIR(bf, kp, kp + 8);
            sim[st] = __builtin_amdgcn_wmma_f32_16x16x32_bf16(false, qf[t].h, false, bf.h,
                                                              (short)0, sim[st], false, false);
        }
    }

    // row max (padded K rows are zero => sim=0 there; any max constant is valid)
    float rmax[8];
#pragma unroll
    for (int r = 0; r < 8; r++) {
        float mx = -3.0e38f;
#pragma unroll
        for (int st = 0; st < ST; st++) mx = fmaxf(mx, sim[st][r]);
        mx = fmaxf(mx, __shfl_xor(mx, 1, 32));
        mx = fmaxf(mx, __shfl_xor(mx, 2, 32));
        mx = fmaxf(mx, __shfl_xor(mx, 4, 32));
        mx = fmaxf(mx, __shfl_xor(mx, 8, 32));
        rmax[r] = mx;
    }

    v8f oacc[4];
#pragma unroll
    for (int j = 0; j < 4; j++)
#pragma unroll
        for (int e = 0; e < 8; e++) oacc[j][e] = 0.f;
    float lsum[8];
#pragma unroll
    for (int r = 0; r < 8; r++) lsum[r] = 0.f;

    const __hip_bfloat16* vbase = Vp + (long)b * Spad * 512;

#pragma unroll
    for (int c2 = 0; c2 < ST / 2; c2++) {
        // stage P chunk (16 x 32) in C-layout -> LDS
#pragma unroll
        for (int half = 0; half < 2; half++) {
            int st = 2 * c2 + half;
            int scol = st * 16 + m;
#pragma unroll
            for (int r = 0; r < 8; r++) {
                float p = __expf((sim[st][r] - rmax[r]) * scale);
                if (scol >= Sact) p = 0.f;
                lsum[r] += p;
                pst[wid][sel8 + r][half * 16 + m] = __float2bfloat16(p);
            }
        }
        __syncthreads();
        // re-read as A fragment (16 x 32)
        Frag pa;
        LOAD_PAIR(pa, &pst[wid][m][kb], &pst[wid][m][kb + 8]);
        // oacc += P_chunk @ V_chunk   (V fragment-packed: 2x b128 per lane)
#pragma unroll
        for (int j = 0; j < 4; j++) {
            Frag vb;
            const __hip_bfloat16* Vblk =
                vbase + (((long)c2 * 32 + (h << 2) + j) << 9) + (lane << 4);
            LOAD_PAIR(vb, Vblk, Vblk + 8);
            oacc[j] = __builtin_amdgcn_wmma_f32_16x16x32_bf16(false, pa.h, false, vb.h,
                                                              (short)0, oacc[j], false, false);
        }
        __syncthreads();
    }

    float rinv[8];
#pragma unroll
    for (int r = 0; r < 8; r++) {
        float s = lsum[r];
        s += __shfl_xor(s, 1, 32);
        s += __shfl_xor(s, 2, 32);
        s += __shfl_xor(s, 4, 32);
        s += __shfl_xor(s, 8, 32);
        rinv[r] = 1.0f / s;
    }

    long rowBase = (long)b * 4096 + n0 + sel8;
#pragma unroll
    for (int j = 0; j < 4; j++)
#pragma unroll
        for (int r = 0; r < 8; r++)
            O[(rowBase + r) * 512 + h * 64 + j * 16 + m] =
                __float2bfloat16(oacc[j][r] * rinv[r]);
}

// ---------------------------------------------------------------------------
extern "C" void kernel_launch(void* const* d_in, const int* in_sizes, int n_in,
                              void* d_out, int out_size, void* d_ws, size_t ws_size,
                              hipStream_t stream) {
    (void)in_sizes; (void)n_in; (void)out_size; (void)ws_size;
    const float* x   = (const float*)d_in[0];
    const float* ctx = (const float*)d_in[1];
    const float* geo = (const float*)d_in[2];
    const float* g1 = (const float*)d_in[3],  *b1 = (const float*)d_in[4];
    const float* g2 = (const float*)d_in[5],  *b2 = (const float*)d_in[6];
    const float* g3 = (const float*)d_in[7],  *b3 = (const float*)d_in[8];
    const float* g4 = (const float*)d_in[9],  *b4 = (const float*)d_in[10];
    const float* cWq = (const float*)d_in[11], *cWk = (const float*)d_in[12];
    const float* cWv = (const float*)d_in[13], *cWo = (const float*)d_in[14];
    const float* cbo = (const float*)d_in[15];
    const float* gWq = (const float*)d_in[16], *gWk = (const float*)d_in[17];
    const float* gWv = (const float*)d_in[18], *gWo = (const float*)d_in[19];
    const float* gbo = (const float*)d_in[20];
    const float* W1  = (const float*)d_in[21], *bf1 = (const float*)d_in[22];
    const float* W2  = (const float*)d_in[23], *bf2 = (const float*)d_in[24];

    typedef __hip_bfloat16 bf;
    const long NT = 16384;                   // B*N
    char* base = (char*)d_ws;
    size_t cur = 0;
    auto alloc = [&](size_t bytes) { void* p = base + cur; cur = (cur + bytes + 255) & ~(size_t)255; return p; };

    float* T  = (float*)alloc(NT * 512 * 4);          // residual stream f32
    bf*    LN = (bf*)   alloc(NT * 512 * 2);          // layernorm output bf16
    char*  BIG = (char*)alloc(NT * 2048 * 2);         // aliased region (67MB)
    float* XT = (float*)(BIG);                         // transposed x (early only)
    bf* QB = (bf*)(BIG);                               // Q        (16.8MB)
    bf* OB = (bf*)(BIG + NT * 512 * 2);                // attn out (16.8MB)
    bf* KP = (bf*)(BIG + 2 * NT * 512 * 2);            // K padded row-major (1MB)
    bf* VP = (bf*)(BIG + 2 * NT * 512 * 2 + 4L * 256 * 512 * 2);  // V packed
    bf* HB = (bf*)(BIG);                               // FFN hidden (reuses all)

    bf* wq_c = (bf*)alloc(512L * 512 * 2);
    bf* wk_c = (bf*)alloc(768L * 512 * 2);
    bf* wv_c = (bf*)alloc(768L * 512 * 2);
    bf* wo_c = (bf*)alloc(512L * 512 * 2);
    bf* wq_g = (bf*)alloc(512L * 512 * 2);
    bf* wk_g = (bf*)alloc(512L * 512 * 2);
    bf* wv_g = (bf*)alloc(512L * 512 * 2);
    bf* wo_g = (bf*)alloc(512L * 512 * 2);
    bf* w1b  = (bf*)alloc(512L * 2048 * 2);
    bf* w2b  = (bf*)alloc(2048L * 512 * 2);
    bf* ctxb = (bf*)alloc(4L * 77 * 768 * 2);
    bf* geob = (bf*)alloc(4L * 256 * 512 * 2);

    auto pack = [&](const float* src, bf* dst, int K, int N) {
        k_pack<<<dim3(1024), dim3(256), 0, stream>>>(src, dst, K, N);
    };
    pack(cWq, wq_c, 512, 512);   pack(cWk, wk_c, 768, 512);
    pack(cWv, wv_c, 768, 512);   pack(cWo, wo_c, 512, 512);
    pack(gWq, wq_g, 512, 512);   pack(gWk, wk_g, 512, 512);
    pack(gWv, wv_g, 512, 512);   pack(gWo, wo_g, 512, 512);
    pack(W1,  w1b,  512, 2048);  pack(W2,  w2b,  2048, 512);
    k_conv<<<dim3(1024), dim3(256), 0, stream>>>(ctx, ctxb, 4L * 77 * 768);
    k_conv<<<dim3(1024), dim3(256), 0, stream>>>(geo, geob, 4L * 256 * 512);

    // t = x^T + LN1(x^T): coalesced transpose then row-wise LN
    k_transpose<<<dim3(128, 16, 4), 256, 0, stream>>>(x, XT, 512, 4096);
    k_ln1f<<<dim3(2048), 256, 0, stream>>>(XT, g1, b1, T);

    // ---------------- context cross-attention ----------------
    k_ln<<<dim3(2048), 256, 0, stream>>>(T, g2, b2, LN);
    k_gemm<<<dim3(64, 8, 1), 256, 0, stream>>>(LN, 512, 0, wq_c, nullptr,
                                               16384, 512, 512, QB, nullptr, 512, 0, 1);
    k_zero_bf16<<<dim3(512), 256, 0, stream>>>(KP, 4L * 256 * 512);
    k_zero_bf16<<<dim3(512), 256, 0, stream>>>(VP, 4L * 256 * 512);
    k_gemm<<<dim3(1, 8, 4), 256, 0, stream>>>(ctxb, 768, 77L * 768, wk_c, nullptr,
                                              77, 512, 768, KP, nullptr, 512, 96L * 512, 1);
    k_gemm<<<dim3(1, 8, 4), 256, 0, stream>>>(ctxb, 768, 77L * 768, wv_c, nullptr,
                                              77, 512, 768, VP, nullptr, 512, 96L * 512, 4);
    k_attn<6><<<dim3(32, 8, 4), 256, 0, stream>>>(QB, KP, VP, OB, 77, 0.125f);
    k_gemm<<<dim3(64, 8, 1), 256, 0, stream>>>(OB, 512, 0, wo_c, cbo,
                                               16384, 512, 512, nullptr, T, 512, 0, 3);

    // ---------------- geometry cross-attention ----------------
    k_ln<<<dim3(2048), 256, 0, stream>>>(T, g3, b3, LN);
    k_gemm<<<dim3(64, 8, 1), 256, 0, stream>>>(LN, 512, 0, wq_g, nullptr,
                                               16384, 512, 512, QB, nullptr, 512, 0, 1);
    k_gemm<<<dim3(1, 8, 4), 256, 0, stream>>>(geob, 512, 256L * 512, wk_g, nullptr,
                                              256, 512, 512, KP, nullptr, 512, 256L * 512, 1);
    k_gemm<<<dim3(1, 8, 4), 256, 0, stream>>>(geob, 512, 256L * 512, wv_g, nullptr,
                                              256, 512, 512, VP, nullptr, 512, 256L * 512, 4);
    k_attn<16><<<dim3(32, 8, 4), 256, 0, stream>>>(QB, KP, VP, OB, 256, 0.125f);
    k_gemm<<<dim3(64, 8, 1), 256, 0, stream>>>(OB, 512, 0, wo_g, gbo,
                                               16384, 512, 512, nullptr, T, 512, 0, 3);

    // ---------------- FFN ----------------
    k_ln<<<dim3(2048), 256, 0, stream>>>(T, g4, b4, LN);
    k_gemm<<<dim3(64, 32, 1), 256, 0, stream>>>(LN, 512, 0, w1b, bf1,
                                                16384, 2048, 512, HB, nullptr, 2048, 0, 2);
    k_gemm<<<dim3(64, 8, 1), 256, 0, stream>>>(HB, 2048, 0, w2b, bf2,
                                               16384, 512, 2048, nullptr, T, 512, 0, 3);

    // t -> (B, C, 64, 64): coalesced transpose back
    k_transpose<<<dim3(16, 128, 4), 256, 0, stream>>>(T, (float*)d_out, 4096, 512);
}
